// EdgeUpdate_16836271801104
// MI455X (gfx1250) — matrix operand order repaired and checked by reading, not compiled, not run on hardware
//
#include <hip/hip_runtime.h>

// ---------------------------------------------------------------------------
// EdgeUpdate on MI455X (gfx1250): all GEMMs via v_wmma_f32_16x16x32_bf16.
//
//   g1(z) = rho(z)+rho(-z) = (relu(t+b1)+relu(b1-t)) @ w2 + 2*b2,  t = z@w1
//   out_e = (edge_attr_e @ W) * ( relu([g1_src|g1_dst] @ phi_w1 + pb1) @ phi_w2 + pb2 )
// ---------------------------------------------------------------------------

typedef __attribute__((ext_vector_type(16))) __bf16 v16bf;
typedef __attribute__((ext_vector_type(8)))  __bf16 v8bf;
typedef __attribute__((ext_vector_type(8)))  float  v8f;

#define DFEAT 128
#define LDSW  136   // padded LDS row stride (bf16 elems) = 272B, 16B aligned

// workspace layout, offsets in bf16 elements
#define OFS_RW1 0
#define OFS_RW2 16384
#define OFS_PW1 32768        // phi_w1 is 256x128 -> 64 tiles
#define OFS_PW2 65536
#define OFS_WW  81920
#define OFS_G   98304        // g1 table: n_nodes x 128 bf16
#define N_WELEM 98304

union AFrag { v16bf v; v8bf h[2]; __bf16 e[16]; };

// A-matrix 16x32 bf16 fragment (ISA 7.12.2): lane(m = lane&15, half = lane>>4)
// holds K = {8*half .. 8*half+7} and {16+8*half .. 23+8*half} of row m.
__device__ __forceinline__ v16bf load_a_bf16(const __bf16* rowp, int k0, int half) {
  AFrag f;
  const __bf16* p = rowp + k0 + half * 8;
  f.h[0] = *(const v8bf*)(p);
  f.h[1] = *(const v8bf*)(p + 16);
  return f.v;
}

__device__ __forceinline__ v16bf load_a_f32(const float* rowp, int k0, int half) {
  AFrag f;
  const float* p = rowp + k0 + half * 8;
#pragma unroll
  for (int i = 0; i < 8; ++i) {
    f.e[i]     = (__bf16)p[i];
    f.e[8 + i] = (__bf16)p[16 + i];
  }
  return f.v;
}

// B fragments are pre-swizzled by prep_weights: tile (kc,ct) = 32 lanes x 16 bf16.
__device__ __forceinline__ v16bf bfrag(const __bf16* mat, int kc, int ct, int lane) {
  return ((const v16bf*)mat)[(size_t)((kc * 8 + ct) * 32 + lane)];
}

__device__ __forceinline__ v8f wmma_bf16(v16bf a, v16bf b, v8f c) {
  return __builtin_amdgcn_wmma_f32_16x16x32_bf16(false, a, false, b, (short)0, c,
                                                 false, false);
}

// ---------------------------------------------------------------------------
// Weight prep: f32 [K x 128] row-major -> fragment-ordered bf16.
// B 32x16 layout: lane (n = lane&15, halfB = lane>>4), element j2 in [0,16)
// holds B[32*kc + 8*halfB + (j2&7) + 16*(j2>>3)][16*ct + n].
// ---------------------------------------------------------------------------
__global__ void __launch_bounds__(256)
prep_weights(const float* __restrict__ rw1, const float* __restrict__ rw2,
             const float* __restrict__ pw1, const float* __restrict__ pw2,
             const float* __restrict__ ww, __bf16* __restrict__ ws) {
  int id = blockIdx.x * blockDim.x + threadIdx.x;
  if (id >= N_WELEM) return;
  const float* src;
  int local;
  if      (id < OFS_RW2) { src = rw1; local = id; }
  else if (id < OFS_PW1) { src = rw2; local = id - OFS_RW2; }
  else if (id < OFS_PW2) { src = pw1; local = id - OFS_PW1; }
  else if (id < OFS_WW)  { src = pw2; local = id - OFS_PW2; }
  else                   { src = ww;  local = id - OFS_WW;  }
  int tile   = local >> 9;      // 512 bf16 per 32x16 tile
  int within = local & 511;
  int lane   = within >> 4;
  int j2     = within & 15;
  int ct     = tile & 7;
  int kc     = tile >> 3;
  int n      = lane & 15;
  int halfB  = lane >> 4;
  int k      = kc * 32 + halfB * 8 + (j2 & 7) + (j2 >> 3) * 16;
  ws[id] = (__bf16)src[(size_t)k * DFEAT + ct * 16 + n];
}

// ---------------------------------------------------------------------------
// Node stage: g1 for all nodes.  One wave per 16 nodes, 8 waves/block.
// ---------------------------------------------------------------------------
__global__ void __launch_bounds__(256)
node_g1(const float* __restrict__ x, const float* __restrict__ b1,
        const float* __restrict__ b2, const __bf16* __restrict__ ws,
        __bf16* __restrict__ g, int n_nodes) {
  __shared__ __align__(16) __bf16 lds[8][16 * LDSW];
  const int lane = threadIdx.x & 31;
  const int wave = threadIdx.x >> 5;
  const int m    = lane & 15;          // A row / D column index
  const int half = lane >> 4;
  const int rowBase = (blockIdx.x * 8 + wave) * 16;

  int row = rowBase + m;
  if (row >= n_nodes) row = n_nodes - 1;        // clamp tail loads
  const float* xp = x + (size_t)row * DFEAT;

  v16bf a[4];
#pragma unroll
  for (int kc = 0; kc < 4; ++kc) a[kc] = load_a_f32(xp, kc * 32, half);

  const __bf16* w1 = ws + OFS_RW1;
  __bf16* lrow = &lds[wave][0];

#pragma unroll
  for (int ct = 0; ct < 8; ++ct) {              // t = x@w1 ; u = relu(t+b1)+relu(b1-t)
    v8f acc = {};
#pragma unroll
    for (int kc = 0; kc < 4; ++kc)
      acc = wmma_bf16(a[kc], bfrag(w1, kc, ct, lane), acc);
    float bv = b1[ct * 16 + m];
#pragma unroll
    for (int r = 0; r < 8; ++r) {
      float t = acc[r];
      float u = fmaxf(t + bv, 0.0f) + fmaxf(bv - t, 0.0f);
      lrow[(r + half * 8) * LDSW + ct * 16 + m] = (__bf16)u;
    }
  }

  asm volatile("s_wait_dscnt 0" ::: "memory");   // LDS transpose barrier (per-wave)

  v16bf ua[4];
#pragma unroll
  for (int kc = 0; kc < 4; ++kc)
    ua[kc] = load_a_bf16(lrow + m * LDSW, kc * 32, half);

  const __bf16* w2 = ws + OFS_RW2;
#pragma unroll
  for (int ct = 0; ct < 8; ++ct) {              // g1 = u@w2 + 2*b2
    v8f acc = {};
#pragma unroll
    for (int kc = 0; kc < 4; ++kc)
      acc = wmma_bf16(ua[kc], bfrag(w2, kc, ct, lane), acc);
    float bv = 2.0f * b2[ct * 16 + m];
#pragma unroll
    for (int r = 0; r < 8; ++r) {
      int orow = rowBase + r + half * 8;
      if (orow < n_nodes)
        g[(size_t)orow * DFEAT + ct * 16 + m] = (__bf16)(acc[r] + bv);
    }
  }
}

// ---------------------------------------------------------------------------
// Edge stage: one wave per 16 edges, 8 waves/block.
// ---------------------------------------------------------------------------
__global__ void __launch_bounds__(256)
edge_update(const float* __restrict__ ea, const int* __restrict__ ei,
            const float* __restrict__ pb1, const float* __restrict__ pb2,
            const __bf16* __restrict__ ws, const __bf16* __restrict__ g,
            float* __restrict__ out, int n_edges) {
  __shared__ __align__(16) __bf16 lds[8][16 * LDSW];
  const int lane = threadIdx.x & 31;
  const int wave = threadIdx.x >> 5;
  const int m    = lane & 15;
  const int half = lane >> 4;
  const int eBase = (blockIdx.x * 8 + wave) * 16;

  int e = eBase + m;
  if (e >= n_edges) e = n_edges - 1;
  const int srcN = ei[e];                 // edge_index[0][e]
  const int dstN = ei[n_edges + e];       // edge_index[1][e]
  const __bf16* gs = g + (size_t)srcN * DFEAT;   // L2-resident gathers
  const __bf16* gd = g + (size_t)dstN * DFEAT;

  v16bf as[4], adf[4];
#pragma unroll
  for (int kc = 0; kc < 4; ++kc) {
    as[kc]  = load_a_bf16(gs, kc * 32, half);
    adf[kc] = load_a_bf16(gd, kc * 32, half);
  }

  const __bf16* w1 = ws + OFS_PW1;       // 256x128 -> kc 0..3 (src), 4..7 (dst)
  __bf16* lrow = &lds[wave][0];

#pragma unroll
  for (int ct = 0; ct < 8; ++ct) {       // u1 = relu([gs|gd]@phi_w1 + pb1)
    v8f acc = {};
#pragma unroll
    for (int kc = 0; kc < 4; ++kc)
      acc = wmma_bf16(as[kc], bfrag(w1, kc, ct, lane), acc);
#pragma unroll
    for (int kc = 0; kc < 4; ++kc)
      acc = wmma_bf16(adf[kc], bfrag(w1, kc + 4, ct, lane), acc);
    float bv = pb1[ct * 16 + m];
#pragma unroll
    for (int r = 0; r < 8; ++r)
      lrow[(r + half * 8) * LDSW + ct * 16 + m] = (__bf16)fmaxf(acc[r] + bv, 0.0f);
  }

  asm volatile("s_wait_dscnt 0" ::: "memory");

  v16bf ua[4], aw[4];
#pragma unroll
  for (int kc = 0; kc < 4; ++kc)
    ua[kc] = load_a_bf16(lrow + m * LDSW, kc * 32, half);
  const float* eap = ea + (size_t)e * DFEAT;     // streaming edge_attr, cvt to bf16
#pragma unroll
  for (int kc = 0; kc < 4; ++kc)
    aw[kc] = load_a_f32(eap, kc * 32, half);

  const __bf16* w2 = ws + OFS_PW2;
  const __bf16* wW = ws + OFS_WW;
#pragma unroll
  for (int ct = 0; ct < 8; ++ct) {       // out = (ea@W) * (u1@phi_w2 + pb2)
    v8f pacc = {};
    v8f wacc = {};
#pragma unroll
    for (int kc = 0; kc < 4; ++kc) {
      pacc = wmma_bf16(ua[kc], bfrag(w2, kc, ct, lane), pacc);
      wacc = wmma_bf16(aw[kc], bfrag(wW, kc, ct, lane), wacc);
    }
    float bv = pb2[ct * 16 + m];
#pragma unroll
    for (int r = 0; r < 8; ++r) {
      int oe = eBase + r + half * 8;
      if (oe < n_edges)
        out[(size_t)oe * DFEAT + ct * 16 + m] = wacc[r] * (pacc[r] + bv);
    }
  }
}

// ---------------------------------------------------------------------------
extern "C" void kernel_launch(void* const* d_in, const int* in_sizes, int n_in,
                              void* d_out, int out_size, void* d_ws, size_t ws_size,
                              hipStream_t stream) {
  const float* x      = (const float*)d_in[0];
  const int*   ei     = (const int*)  d_in[1];
  const float* ea     = (const float*)d_in[2];
  const float* rho_w1 = (const float*)d_in[3];
  const float* rho_b1 = (const float*)d_in[4];
  const float* rho_w2 = (const float*)d_in[5];
  const float* rho_b2 = (const float*)d_in[6];
  const float* phi_w1 = (const float*)d_in[7];
  const float* phi_b1 = (const float*)d_in[8];
  const float* phi_w2 = (const float*)d_in[9];
  const float* phi_b2 = (const float*)d_in[10];
  const float* W_w    = (const float*)d_in[11];
  float* out = (float*)d_out;

  const int n_nodes = in_sizes[0] / DFEAT;
  const int n_edges = in_sizes[2] / DFEAT;

  __bf16* ws = (__bf16*)d_ws;
  __bf16* g  = ws + OFS_G;

  // 1) weights -> fragment-ordered bf16 in workspace
  prep_weights<<<(N_WELEM + 255) / 256, 256, 0, stream>>>(
      rho_w1, rho_w2, phi_w1, phi_w2, W_w, ws);

  // 2) g1(x) for all nodes (128 nodes / block)
  node_g1<<<(n_nodes + 127) / 128, 256, 0, stream>>>(
      x, rho_b1, rho_b2, ws, g, n_nodes);

  // 3) fused edge update (128 edges / block)
  edge_update<<<(n_edges + 127) / 128, 256, 0, stream>>>(
      ea, ei, phi_b1, phi_b2, ws, g, out, n_edges);
}
